// SharedAttnProcessor_59785944760509
// MI455X (gfx1250) — matrix operand order repaired and checked by reading, not compiled, not run on hardware
//
#include <hip/hip_runtime.h>
#include <hip/hip_bf16.h>

// ---------------------------------------------------------------- types
typedef __attribute__((ext_vector_type(16))) __bf16 v16bf;
typedef __attribute__((ext_vector_type(8)))  float  v8f;
typedef __attribute__((ext_vector_type(8)))  __bf16 bf16x8;   // 16B, for b128 ld/st
typedef __attribute__((ext_vector_type(4)))  float  f32x4;    // 16B

// problem constants (from reference setup_inputs)
#define AB  2
#define AS  1024
#define AC  640
#define AH  8
#define AD  80
#define ANR 4
#define AM  (AB*AS)          // 2048 rows for the GEMMs
#define AKV ((1+ANR)*AS)     // 5120 extended kv length

// ---------------------------------------------------------------- WMMA helpers
__device__ __forceinline__ v8f wmma_bf16(v16bf a, v16bf b, v8f c) {
  // v_wmma_f32_16x16x32_bf16: D = A(16x32) * B(32x16) + C
  return __builtin_amdgcn_wmma_f32_16x16x32_bf16(
      /*neg_a=*/false, a, /*neg_b=*/false, b,
      /*c_mod=*/(short)0, c, /*reuse_a=*/false, /*reuse_b=*/false);
}

// A fragment (16x32 bf16): lane holds row (lane&15); lanes<16 get K runs
// [koff..koff+7] and [koff+16..koff+23]; lanes>=16 get the +8 runs.
__device__ __forceinline__ v16bf load_A(const __bf16* base, int stride,
                                        int row0, int koff, int lane) {
  int m  = lane & 15;
  int hi = lane >> 4;
  const __bf16* p = base + (row0 + m) * stride + koff + hi * 8;
  v16bf a;
#pragma unroll
  for (int j = 0; j < 8; ++j) a[j] = p[j];
#pragma unroll
  for (int j = 0; j < 8; ++j) a[8 + j] = p[16 + j];
  return a;
}

// B fragment (32x16 bf16): lane holds column (lane&15); lanes<16 get K runs
// [koff..koff+7] and [koff+16..koff+23]; lanes>=16 get the +8 runs.
// Source must be laid out [col][k] (k contiguous).
__device__ __forceinline__ v16bf load_B(const __bf16* base, int stride,
                                        int col0, int koff, int lane) {
  int n  = lane & 15;
  int hi = lane >> 4;
  const __bf16* p = base + (col0 + n) * stride + koff + hi * 8;
  v16bf b;
#pragma unroll
  for (int j = 0; j < 8; ++j) b[j] = p[j];
#pragma unroll
  for (int j = 0; j < 8; ++j) b[8 + j] = p[16 + j];
  return b;
}

// ---------------------------------------------------------------- GEMM (proj / out)
// MODE 0: A=f32, out=bf16, no bias (projections)
// MODE 1: A=bf16, out=f32 + bias   (final output projection)
template <int MODE>
__global__ void __launch_bounds__(128, 4)
gemm_wmma(const void* __restrict__ Ap, const float* __restrict__ W,
          void* __restrict__ Cp, const float* __restrict__ bias,
          int M, int K, int N) {
  __shared__ __attribute__((aligned(16))) __bf16 sA[32 * 34];   // [row][k]
  __shared__ __attribute__((aligned(16))) __bf16 sWt[64 * 34];  // [col][k] (W^T)

  int tid  = threadIdx.x;
  int lane = tid & 31, wid = tid >> 5;
  int m0 = blockIdx.x * 32, n0 = blockIdx.y * 64;
  int wm = wid & 1, wn = wid >> 1;

  v8f c0 = {}, c1 = {};
#pragma unroll 1
  for (int kc = 0; kc < K; kc += 32) {
    // stage A tile (32 x 32) as bf16
#pragma unroll
    for (int j = 0; j < 8; ++j) {
      int e = tid + 128 * j;
      int r = e >> 5, kk = e & 31;
      float v;
      if (MODE == 0) v = ((const float*)Ap)[(size_t)(m0 + r) * K + kc + kk];
      else           v = (float)((const __bf16*)Ap)[(size_t)(m0 + r) * K + kc + kk];
      sA[r * 34 + kk] = (__bf16)v;
    }
    // stage W tile (32 x 64) transposed -> [col][k]
#pragma unroll
    for (int j = 0; j < 16; ++j) {
      int e = tid + 128 * j;
      int kk = e >> 6, nn = e & 63;
      sWt[nn * 34 + kk] = (__bf16)W[(size_t)(kc + kk) * N + n0 + nn];
    }
    __syncthreads();

    v16bf a  = load_A(sA, 34, wm * 16, 0, lane);
    v16bf b0 = load_B(sWt, 34, wn * 32, 0, lane);
    v16bf b1 = load_B(sWt, 34, wn * 32 + 16, 0, lane);
    c0 = wmma_bf16(a, b0, c0);
    c1 = wmma_bf16(a, b1, c1);
    __syncthreads();
  }

  int nn = lane & 15, hi = lane >> 4;
#pragma unroll
  for (int i = 0; i < 8; ++i) {
    int row = m0 + wm * 16 + i + 8 * hi;
    int col = n0 + wn * 32 + nn;
    if (MODE == 0) {
      ((__bf16*)Cp)[(size_t)row * N + col]      = (__bf16)c0[i];
      ((__bf16*)Cp)[(size_t)row * N + col + 16] = (__bf16)c1[i];
    } else {
      ((float*)Cp)[(size_t)row * N + col]      = c0[i] + bias[col];
      ((float*)Cp)[(size_t)row * N + col + 16] = c1[i] + bias[col + 16];
    }
  }
}

// ---------------------------------------------------------------- AdaIN statistics
__global__ void __launch_bounds__(256)
style_stats(const __bf16* __restrict__ Vb, float* __restrict__ smean,
            float* __restrict__ sstd) {
  __shared__ float rs[256], rq[256];
  int bc = blockIdx.x;               // b*C + c
  int b = bc / AC, c = bc % AC;
  float sum = 0.f, sq = 0.f;
  for (int s = threadIdx.x; s < AS; s += 256) {
    float v = (float)Vb[((size_t)(b * AS + s)) * AC + c];
    sum += v; sq += v * v;
  }
  rs[threadIdx.x] = sum; rq[threadIdx.x] = sq;
  __syncthreads();
  for (int o = 128; o > 0; o >>= 1) {
    if (threadIdx.x < o) { rs[threadIdx.x] += rs[threadIdx.x + o];
                           rq[threadIdx.x] += rq[threadIdx.x + o]; }
    __syncthreads();
  }
  if (threadIdx.x == 0) {
    float m   = rs[0] / (float)AS;
    float var = (rq[0] - rs[0] * rs[0] / (float)AS) / (float)(AS - 1);
    smean[bc] = m;
    sstd[bc]  = sqrtf(fmaxf(var, 0.f)) + 1e-5f;
  }
}

// content stats over each ref-V; emit affine (scale, shift):  rv' = rv*scale + shift
__global__ void __launch_bounds__(256)
content_stats(const float* __restrict__ rV, const float* __restrict__ smean,
              const float* __restrict__ sstd, float* __restrict__ scl,
              float* __restrict__ shf) {
  __shared__ float rs[256], rq[256];
  int idx = blockIdx.x;              // (b*NR + r)*C + c
  int c  = idx % AC;
  int br = idx / AC;                 // b*NR + r
  int b  = br / ANR;
  float sum = 0.f, sq = 0.f;
  for (int s = threadIdx.x; s < AS; s += 256) {
    float v = rV[((size_t)br * AS + s) * AC + c];
    sum += v; sq += v * v;
  }
  rs[threadIdx.x] = sum; rq[threadIdx.x] = sq;
  __syncthreads();
  for (int o = 128; o > 0; o >>= 1) {
    if (threadIdx.x < o) { rs[threadIdx.x] += rs[threadIdx.x + o];
                           rq[threadIdx.x] += rq[threadIdx.x + o]; }
    __syncthreads();
  }
  if (threadIdx.x == 0) {
    float cm  = rs[0] / (float)AS;
    float var = (rq[0] - rs[0] * rs[0] / (float)AS) / (float)(AS - 1);
    float cs  = sqrtf(fmaxf(var, 0.f)) + 1e-5f;
    float k   = sstd[b * AC + c] / cs;
    scl[idx] = k;
    shf[idx] = smean[b * AC + c] - cm * k;
  }
}

// ---------------------------------------------------------------- flash attention
// grid: 256 blocks = (b,h) * 16 q-tiles of 64 rows; 4 waves, 16 q rows each.
// __launch_bounds__(128, 4): force <=256 VGPRs (kills s_set_vgpr_msb thrash).
__global__ void __launch_bounds__(128, 4)
attn_wmma(const __bf16* __restrict__ Qb, const __bf16* __restrict__ Kb,
          const __bf16* __restrict__ Vb, const float* __restrict__ rK,
          const float* __restrict__ rV, const float* __restrict__ scl,
          const float* __restrict__ shf, __bf16* __restrict__ Ob) {
  __shared__ __attribute__((aligned(16))) __bf16 sK[32 * 104];  // [kv][d0..95]
  __shared__ __attribute__((aligned(16))) __bf16 sVt[80 * 34];  // [d][kv]
  __shared__ __attribute__((aligned(16))) __bf16 sP[4 * 16 * 36];

  int tid = threadIdx.x, lane = tid & 31, wid = tid >> 5;
  int qt = blockIdx.x & 15;
  int bh = blockIdx.x >> 4;
  int b = bh >> 3, h = bh & 7;
  int hi = lane >> 4, mn = lane & 15;

  // Q A-fragments, D=80 zero-padded to 96 (3 K-chunks of 32)
  v16bf aq[3];
  {
    int qr = qt * 64 + wid * 16 + mn;
    const __bf16* qp = Qb + ((size_t)(b * AS + qr)) * AC + h * AD;
#pragma unroll
    for (int kc = 0; kc < 3; ++kc) {
      int d0 = kc * 32 + hi * 8;
#pragma unroll
      for (int j = 0; j < 8; ++j) {
        int d  = d0 + j;
        int d2 = d0 + 16 + j;
        aq[kc][j]     = (d  < AD) ? qp[d]  : (__bf16)0.0f;
        aq[kc][8 + j] = (d2 < AD) ? qp[d2] : (__bf16)0.0f;
      }
    }
  }

  v8f o[5] = {};
  float mrun[8], lrun[8];
#pragma unroll
  for (int i = 0; i < 8; ++i) { mrun[i] = -1e30f; lrun[i] = 0.f; }
  const float sc = 0.11180339887498949f;  // 80^-0.5

#pragma unroll 1
  for (int kv0 = 0; kv0 < AKV; kv0 += 32) {
    bool isSelf = kv0 < AS;
    int ridx  = isSelf ? 0 : (kv0 - AS) / AS;
    int sbase = isSelf ? kv0 : (kv0 - AS) % AS;

    // prefetch the next chunk's rows (global_prefetch_b8); one row per lane
    {
      int nkv = kv0 + 32;
      if (nkv < AKV) {
        bool ns = nkv < AS;
        int nr  = ns ? 0 : (nkv - AS) / AS;
        int nsb = (ns ? nkv : (nkv - AS) % AS) + (tid & 31);
        const void* pk = ns
            ? (const void*)(Kb + ((size_t)(b * AS + nsb)) * AC + h * AD)
            : (const void*)(rK + ((size_t)(b * ANR + nr) * AS + nsb) * AC + h * AD);
        const void* pv = ns
            ? (const void*)(Vb + ((size_t)(b * AS + nsb)) * AC + h * AD)
            : (const void*)(rV + ((size_t)(b * ANR + nr) * AS + nsb) * AC + h * AD);
        __builtin_prefetch(pk, 0, 1);
        __builtin_prefetch(pv, 0, 1);
      }
    }

    // ---- stage K tile [32][96] (d>=80 zero-padded), 8-element vector chunks
    // 32 rows * 12 chunks = 384 = 128 threads * 3
#pragma unroll
    for (int j = 0; j < 3; ++j) {
      int cidx = tid + 128 * j;
      int r = cidx / 12, db = (cidx % 12) * 8;
      bf16x8 val;
      if (db < AD) {
        if (isSelf) {
          val = *(const bf16x8*)(Kb + ((size_t)(b * AS + sbase + r)) * AC + h * AD + db);
        } else {
          const float* p = rK + ((size_t)(b * ANR + ridx) * AS + sbase + r) * AC + h * AD + db;
          f32x4 f0 = *(const f32x4*)p;
          f32x4 f1 = *(const f32x4*)(p + 4);
#pragma unroll
          for (int t = 0; t < 4; ++t) { val[t] = (__bf16)f0[t]; val[4 + t] = (__bf16)f1[t]; }
        }
      } else {
#pragma unroll
        for (int t = 0; t < 8; ++t) val[t] = (__bf16)0.0f;
      }
      *(bf16x8*)(sK + r * 104 + db) = val;   // 16B-aligned ds_store_b128
    }

    // ---- stage V tile transposed [d][kv] with AdaIN; vector global loads,
    // scattered bf16 LDS stores. 32 rows * 10 chunks = 320 chunks.
#pragma unroll
    for (int j = 0; j < 3; ++j) {
      int cidx = tid + 128 * j;
      if (cidx < 320) {
        int r = cidx / 10, db = (cidx % 10) * 8;
        float f[8];
        if (isSelf) {
          bf16x8 v8 = *(const bf16x8*)(Vb + ((size_t)(b * AS + sbase + r)) * AC + h * AD + db);
#pragma unroll
          for (int t = 0; t < 8; ++t) f[t] = (float)v8[t];
        } else {
          const float* p = rV + ((size_t)(b * ANR + ridx) * AS + sbase + r) * AC + h * AD + db;
          int si = (b * ANR + ridx) * AC + h * AD + db;
          f32x4 v0 = *(const f32x4*)p,          v1 = *(const f32x4*)(p + 4);
          f32x4 k0 = *(const f32x4*)(scl + si), k1 = *(const f32x4*)(scl + si + 4);
          f32x4 s0 = *(const f32x4*)(shf + si), s1 = *(const f32x4*)(shf + si + 4);
#pragma unroll
          for (int t = 0; t < 4; ++t) { f[t] = v0[t] * k0[t] + s0[t];
                                        f[4 + t] = v1[t] * k1[t] + s1[t]; }
        }
#pragma unroll
        for (int t = 0; t < 8; ++t) sVt[(db + t) * 34 + r] = (__bf16)f[t];
      }
    }
    __syncthreads();

    // scores: two 16x16 tiles over this kv chunk, K=96
    v8f s0 = {}, s1 = {};
#pragma unroll
    for (int kc = 0; kc < 3; ++kc) {
      s0 = wmma_bf16(aq[kc], load_B(sK, 104, 0,  kc * 32, lane), s0);
      s1 = wmma_bf16(aq[kc], load_B(sK, 104, 16, kc * 32, lane), s1);
    }

    // online softmax: each score row lives in one 16-lane half
    __bf16* myP = sP + wid * 16 * 36;
#pragma unroll
    for (int i = 0; i < 8; ++i) {
      float t0 = s0[i] * sc, t1 = s1[i] * sc;
      float rm = fmaxf(t0, t1);
#pragma unroll
      for (int off = 1; off < 16; off <<= 1) rm = fmaxf(rm, __shfl_xor(rm, off, 16));
      float mnew  = fmaxf(mrun[i], rm);
      float alpha = __expf(mrun[i] - mnew);
      mrun[i] = mnew;
      float p0 = __expf(t0 - mnew), p1 = __expf(t1 - mnew);
      float rs = p0 + p1;
#pragma unroll
      for (int off = 1; off < 16; off <<= 1) rs += __shfl_xor(rs, off, 16);
      lrun[i] = lrun[i] * alpha + rs;
#pragma unroll
      for (int d = 0; d < 5; ++d) o[d][i] *= alpha;
      int row = i + 8 * hi;
      myP[row * 36 + mn]      = (__bf16)p0;
      myP[row * 36 + 16 + mn] = (__bf16)p1;
    }

    // P(16x32) @ V(32x80): 5 accumulating WMMAs
    v16bf ap = load_A(myP, 36, 0, 0, lane);
#pragma unroll
    for (int d = 0; d < 5; ++d)
      o[d] = wmma_bf16(ap, load_B(sVt, 34, d * 16, 0, lane), o[d]);
    __syncthreads();
  }

  // normalize and write O in [B,S,C] (bf16) for the final projection GEMM
  int qr = qt * 64 + wid * 16;
#pragma unroll
  for (int i = 0; i < 8; ++i) {
    float invl = 1.0f / lrun[i];
    int row = qr + i + 8 * hi;
#pragma unroll
    for (int d = 0; d < 5; ++d)
      Ob[((size_t)(b * AS + row)) * AC + h * AD + d * 16 + mn] =
          (__bf16)(o[d][i] * invl);
  }
}

// ---------------------------------------------------------------- launch
extern "C" void kernel_launch(void* const* d_in, const int* in_sizes, int n_in,
                              void* d_out, int out_size, void* d_ws, size_t ws_size,
                              hipStream_t stream) {
  const float* hs = (const float*)d_in[0];
  const float* rk = (const float*)d_in[1];
  const float* rv = (const float*)d_in[2];
  const float* Wq = (const float*)d_in[3];
  const float* Wk = (const float*)d_in[4];
  const float* Wv = (const float*)d_in[5];
  const float* Wo = (const float*)d_in[6];
  const float* bo = (const float*)d_in[7];
  float* out = (float*)d_out;

  // workspace carve-up (all 256B-aligned chunks)
  const size_t MC = (size_t)AM * AC;   // 1,310,720 elements
  char* w = (char*)d_ws;
  __bf16* Qb = (__bf16*)w; w += MC * sizeof(__bf16);
  __bf16* Kb = (__bf16*)w; w += MC * sizeof(__bf16);
  __bf16* Vb = (__bf16*)w; w += MC * sizeof(__bf16);
  __bf16* Ob = (__bf16*)w; w += MC * sizeof(__bf16);
  float* smean = (float*)w; w += (size_t)AB * AC * sizeof(float);
  float* sstd  = (float*)w; w += (size_t)AB * AC * sizeof(float);
  float* scl   = (float*)w; w += (size_t)AB * ANR * AC * sizeof(float);
  float* shf   = (float*)w; w += (size_t)AB * ANR * AC * sizeof(float);

  dim3 gG(AM / 32, AC / 64), bG(128);
  // projections
  gemm_wmma<0><<<gG, bG, 0, stream>>>(hs, Wq, Qb, nullptr, AM, AC, AC);
  gemm_wmma<0><<<gG, bG, 0, stream>>>(hs, Wk, Kb, nullptr, AM, AC, AC);
  gemm_wmma<0><<<gG, bG, 0, stream>>>(hs, Wv, Vb, nullptr, AM, AC, AC);
  // AdaIN stats -> affine params
  style_stats<<<AB * AC, 256, 0, stream>>>(Vb, smean, sstd);
  content_stats<<<AB * ANR * AC, 256, 0, stream>>>(rv, smean, sstd, scl, shf);
  // flash attention over extended KV
  attn_wmma<<<AB * AH * (AS / 64), 128, 0, stream>>>(Qb, Kb, Vb, rk, rv, scl, shf, Ob);
  // output projection + bias
  gemm_wmma<1><<<gG, bG, 0, stream>>>(Ob, Wo, out, bo, AM, AC, AC);
}
